// NW2_73572789780679
// MI455X (gfx1250) — matrix operand order repaired and checked by reading, not compiled, not run on hardware
//
#include <hip/hip_runtime.h>
#include <hip/hip_bf16.h>

typedef __attribute__((ext_vector_type(16))) _Float16 v16h;
typedef __attribute__((ext_vector_type(8)))  _Float16 v8h;
typedef __attribute__((ext_vector_type(8)))  float    v8f;

union V16 { v16h v; v8h h[2]; };

static constexpr int B_   = 4;
static constexpr int NK_  = 4096;
static constexpr int NQ_  = 4096;
static constexpr int TOK_ = B_ * NK_;   // 16384 tokens per MLP input

// ---------------------------------------------------------------------------
// MLP GEMM:  Y[t,o] = epilogue( X[t,:] . W[o,:] + bias[o] )
//   mode 0: relu     mode 1: bias only    mode 2: (gemm+b) * xsrc[t,o]
// Tile: 128 tokens x 64 outputs per block, K stepped by 32, f16 WMMA.
// ---------------------------------------------------------------------------
__global__ __launch_bounds__(256)
void mlp_gemm(const float* __restrict__ X, const float* __restrict__ W,
              const float* __restrict__ bias, const float* __restrict__ xsrc,
              float* __restrict__ Y, int in_dim, int out_dim, int mode)
{
    __shared__ alignas(16) _Float16 Xh[128 * 40];   // A tile [tok][k], pad 40
    __shared__ alignas(16) _Float16 Wh[64  * 40];   // B^T tile [o][k], pad 40

    const int t    = threadIdx.x;
    const int lane = t & 31;
    const int wave = t >> 5;          // 8 waves = 8 M-subtiles of 16 tokens
    const int hi   = (lane >> 4) & 1;
    const int tok0  = blockIdx.x * 128;
    const int obase = blockIdx.y * 64;

    v8f zero = {0.f,0.f,0.f,0.f,0.f,0.f,0.f,0.f};
    v8f acc[4] = { zero, zero, zero, zero };

    for (int k0 = 0; k0 < in_dim; k0 += 32) {
        {   // stage X tile as f16 (128 x 32)
            const int row = t >> 1, part = (t & 1) * 16;
            const float* xp = X + (size_t)(tok0 + row) * in_dim + k0 + part;
            _Float16* dst = &Xh[row * 40 + part];
            #pragma unroll
            for (int i = 0; i < 16; ++i) dst[i] = (_Float16)xp[i];
        }
        {   // stage W tile as f16 (64 x 32); W rows ARE B^T rows
            const int o = t >> 2, part = (t & 3) * 8;
            const float* wp = W + (size_t)(obase + o) * in_dim + k0 + part;
            _Float16* dst = &Wh[o * 40 + part];
            #pragma unroll
            for (int i = 0; i < 8; ++i) dst[i] = (_Float16)wp[i];
        }
        __syncthreads();

        // A fragment: lane<16 -> K{0..7,16..23}; lane>=16 -> K{8..15,24..31}
        V16 A;
        const int arow = wave * 16 + (lane & 15);
        A.h[0] = *(const v8h*)&Xh[arow * 40 + hi * 8];
        A.h[1] = *(const v8h*)&Xh[arow * 40 + hi * 8 + 16];

        // load ALL B fragments first so WMMAs can issue back-to-back
        V16 Bf[4];
        #pragma unroll
        for (int nt = 0; nt < 4; ++nt) {
            const int brow = nt * 16 + (lane & 15);
            Bf[nt].h[0] = *(const v8h*)&Wh[brow * 40 + hi * 8];
            Bf[nt].h[1] = *(const v8h*)&Wh[brow * 40 + hi * 8 + 16];
        }
        #pragma unroll
        for (int nt = 0; nt < 4; ++nt)
            acc[nt] = __builtin_amdgcn_wmma_f32_16x16x32_f16(
                false, A.v, false, Bf[nt].v, (short)0, acc[nt], false, false);
        __syncthreads();
    }

    // epilogue: C VGPR r -> (M = r + 8*hi, N = lane&15)
    #pragma unroll
    for (int nt = 0; nt < 4; ++nt) {
        const int o  = obase + nt * 16 + (lane & 15);
        const float bo = bias[o];
        #pragma unroll
        for (int r = 0; r < 8; ++r) {
            const int token = tok0 + wave * 16 + r + hi * 8;
            float val = acc[nt][r] + bo;
            if (mode == 0)      val = fmaxf(val, 0.f);
            else if (mode == 2) val *= xsrc[(size_t)token * 64 + o]; // out_dim==64
            Y[(size_t)token * out_dim + o] = val;
        }
    }
}

// ---------------------------------------------------------------------------
// Fused L1-distance attention (flash style, softmax over keys):
//   s[k,q] = sum_d |Ks[k,d]-Qs[q,d]| ;  l = -0.5 s^2 ; online softmax over k
//   O[q,v] = (sum_k p[k,q] V[k,v]) / den[q] * Wo[q,v]
// One block = (b, 64 queries); key tiles of 32; P^T@V via f16 WMMA.
// K tiles: double-buffered in LDS via ASYNC global->LDS DMA (ASYNCcnt).
// V tiles: register double-buffer (needs f32->f16 transpose conversion).
// ---------------------------------------------------------------------------
__global__ __launch_bounds__(256)
void attn_kernel(const float* __restrict__ Ks, const float* __restrict__ Qs,
                 const float* __restrict__ V,  const float* __restrict__ Wo,
                 float* __restrict__ out)
{
    __shared__ alignas(16) float    Qt[64 * 68];      // Q_scale tile [q][d]
    __shared__ alignas(16) float    Kt2[2][32 * 68];  // K tiles (async dest)
    __shared__ alignas(16) float    Lt[32 * 68];      // logits [k][q]
    __shared__ alignas(16) _Float16 Pt[64 * 40];      // P^T [q][k] f16 (A frags)
    __shared__ alignas(16) _Float16 Vt[64 * 40];      // V^T [v][k] f16 (B frags)
    __shared__ float m_s[64], den_s[64], al_s[64], mn_s[64];
    __shared__ float ps_s[4 * 64];                    // partial exp sums

    const int t    = threadIdx.x;
    const int lane = t & 31;
    const int wave = t >> 5;
    const int hi   = (lane >> 4) & 1;
    const int b    = blockIdx.y;
    const int q0   = blockIdx.x * 64;

    // staging indices: each thread owns 8 contiguous d of one k-row
    const int kk = t >> 3;
    const int db = (t & 7) * 8;

    // prologue: async-DMA K tile 0 straight into LDS (no VGPR data path);
    // INST_OFFSET is added to BOTH the LDS and global addresses.
    {
        unsigned lds = (unsigned)(uintptr_t)&Kt2[0][kk * 68 + db];
        const float* gp = Ks + ((size_t)b * NK_ + kk) * 64 + db;
        asm volatile("global_load_async_to_lds_b128 %0, %1, off\n\t"
                     "global_load_async_to_lds_b128 %0, %1, off offset:16"
                     :: "v"(lds), "v"(gp) : "memory");
    }

    {   // stage query tile once (overlaps the async DMA above)
        const int row = t >> 2, part = (t & 3) * 16;
        const float* qp = Qs + ((size_t)b * NQ_ + q0 + row) * 64 + part;
        #pragma unroll
        for (int i = 0; i < 16; i += 4)
            *(float4*)&Qt[row * 68 + part + i] = *(const float4*)&qp[i];
    }
    if (t < 64) { m_s[t] = -3.0e38f; den_s[t] = 0.f; }

    // V tile 0 into registers
    float4 va0, va1;
    {
        const float* vp = V + ((size_t)b * NK_ + kk) * 64 + db;
        va0 = ((const float4*)vp)[0];  va1 = ((const float4*)vp)[1];
    }
    asm volatile("s_wait_asynccnt 0x0" ::: "memory");   // K tile 0 resident

    v8f zero = {0.f,0.f,0.f,0.f,0.f,0.f,0.f,0.f};
    v8f acc0 = zero, acc1 = zero;

    const int sq   = (t & 15) * 4;      // this thread's 4 queries
    const int sk   = (t >> 4) * 2;      // this thread's 2 keys
    const int qsub = wave & 3;          // wave's 16-query subtile
    const int vsub = (wave >> 2) * 2;   // wave's two 16-value subtiles

    for (int it = 0; it < NK_ / 32; ++it) {
        const int k0 = it * 32;
        const float* Kcur = Kt2[it & 1];
        __syncthreads();   // prev iter's reads of Kt2[next]/Vt/Pt all complete

        // kick off async DMA of the NEXT K tile into the other buffer
        if (k0 + 32 < NK_) {
            unsigned lds = (unsigned)(uintptr_t)&Kt2[(it + 1) & 1][kk * 68 + db];
            const float* gp = Ks + ((size_t)b * NK_ + k0 + 32 + kk) * 64 + db;
            asm volatile("global_load_async_to_lds_b128 %0, %1, off\n\t"
                         "global_load_async_to_lds_b128 %0, %1, off offset:16"
                         :: "v"(lds), "v"(gp) : "memory");
        }

        // commit prefetched V tile to LDS transposed as f16
        Vt[(db+0)*40+kk] = (_Float16)va0.x; Vt[(db+1)*40+kk] = (_Float16)va0.y;
        Vt[(db+2)*40+kk] = (_Float16)va0.z; Vt[(db+3)*40+kk] = (_Float16)va0.w;
        Vt[(db+4)*40+kk] = (_Float16)va1.x; Vt[(db+5)*40+kk] = (_Float16)va1.y;
        Vt[(db+6)*40+kk] = (_Float16)va1.z; Vt[(db+7)*40+kk] = (_Float16)va1.w;

        // issue next V tile's global loads; wait lands at next iter's commit
        if (k0 + 32 < NK_) {
            const float* vp = V + ((size_t)b * NK_ + k0 + 32 + kk) * 64 + db;
            va0 = ((const float4*)vp)[0];  va1 = ((const float4*)vp)[1];
        }
        __syncthreads();

        // L1 distances: 8 scores per thread, float4 over d
        float s[8] = {0,0,0,0,0,0,0,0};
        for (int d = 0; d < 64; d += 4) {
            float4 ka = *(const float4*)&Kcur[(sk+0) * 68 + d];
            float4 kb = *(const float4*)&Kcur[(sk+1) * 68 + d];
            #pragma unroll
            for (int j = 0; j < 4; ++j) {
                float4 qv = *(const float4*)&Qt[(sq+j) * 68 + d];
                s[j]   += fabsf(ka.x-qv.x)+fabsf(ka.y-qv.y)+fabsf(ka.z-qv.z)+fabsf(ka.w-qv.w);
                s[4+j] += fabsf(kb.x-qv.x)+fabsf(kb.y-qv.y)+fabsf(kb.z-qv.z)+fabsf(kb.w-qv.w);
            }
        }
        #pragma unroll
        for (int j = 0; j < 4; ++j) {
            Lt[(sk+0) * 68 + sq + j] = -0.5f * s[j]   * s[j];
            Lt[(sk+1) * 68 + sq + j] = -0.5f * s[4+j] * s[4+j];
        }
        __syncthreads();

        // softmax phase A: per-query tile max, running max, alpha (64 threads)
        if (t < 64) {
            const int q = t;
            float mx = -3.0e38f;
            #pragma unroll
            for (int k2 = 0; k2 < 32; ++k2) mx = fmaxf(mx, Lt[k2 * 68 + q]);
            const float mold = m_s[q];
            const float mnew = fmaxf(mold, mx);
            al_s[q] = __expf(mold - mnew);
            mn_s[q] = mnew;
            m_s[q]  = mnew;
        }
        __syncthreads();

        // softmax phase B: exps spread over all 256 threads (8 each)
        {
            const int q  = t & 63;
            const int kg = (t >> 6) * 8;
            const float mnew = mn_s[q];
            float sum = 0.f;
            #pragma unroll
            for (int i = 0; i < 8; ++i) {
                float p = __expf(Lt[(kg + i) * 68 + q] - mnew);
                Pt[q * 40 + kg + i] = (_Float16)p;
                sum += p;
            }
            ps_s[(t >> 6) * 64 + q] = sum;
        }
        __syncthreads();

        // denominator update rides along with the WMMA phase
        if (t < 64)
            den_s[t] = den_s[t] * al_s[t]
                     + ps_s[t] + ps_s[64 + t] + ps_s[128 + t] + ps_s[192 + t];

        // rescale running accumulators, then O += P^T @ V  (two WMMAs/wave)
        #pragma unroll
        for (int r = 0; r < 8; ++r) {
            const float al = al_s[qsub * 16 + r + hi * 8];
            acc0[r] *= al;
            acc1[r] *= al;
        }
        V16 A;
        const int arow = qsub * 16 + (lane & 15);
        A.h[0] = *(const v8h*)&Pt[arow * 40 + hi * 8];
        A.h[1] = *(const v8h*)&Pt[arow * 40 + hi * 8 + 16];
        V16 B0, B1;
        const int brow0 = (vsub + 0) * 16 + (lane & 15);
        const int brow1 = (vsub + 1) * 16 + (lane & 15);
        B0.h[0] = *(const v8h*)&Vt[brow0 * 40 + hi * 8];
        B0.h[1] = *(const v8h*)&Vt[brow0 * 40 + hi * 8 + 16];
        B1.h[0] = *(const v8h*)&Vt[brow1 * 40 + hi * 8];
        B1.h[1] = *(const v8h*)&Vt[brow1 * 40 + hi * 8 + 16];
        acc0 = __builtin_amdgcn_wmma_f32_16x16x32_f16(
            false, A.v, false, B0.v, (short)0, acc0, false, false);
        acc1 = __builtin_amdgcn_wmma_f32_16x16x32_f16(
            false, A.v, false, B1.v, (short)0, acc1, false, false);

        // drain this wave's async DMA (tile it+1) before next loop-top barrier
        asm volatile("s_wait_asynccnt 0x0" ::: "memory");
    }
    __syncthreads();

    // normalize, multiply by Wo, store
    const int n = lane & 15;
    #pragma unroll
    for (int r = 0; r < 8; ++r) {
        const int ql = qsub * 16 + r + hi * 8;
        const float inv = 1.f / den_s[ql];
        const size_t base = ((size_t)b * NQ_ + q0 + ql) * 64;
        const int v0 = (vsub + 0) * 16 + n;
        const int v1 = (vsub + 1) * 16 + n;
        out[base + v0] = acc0[r] * inv * Wo[base + v0];
        out[base + v1] = acc1[r] * inv * Wo[base + v1];
    }
}

// ---------------------------------------------------------------------------
extern "C" void kernel_launch(void* const* d_in, const int* in_sizes, int n_in,
                              void* d_out, int out_size, void* d_ws, size_t ws_size,
                              hipStream_t stream) {
    const float* KEY   = (const float*)d_in[0];
    const float* VALUE = (const float*)d_in[1];
    const float* QUERY = (const float*)d_in[2];
    const float* W1w = (const float*)d_in[3];  const float* W1b = (const float*)d_in[4];
    const float* W2w = (const float*)d_in[5];  const float* W2b = (const float*)d_in[6];
    const float* W3w = (const float*)d_in[7];  const float* W3b = (const float*)d_in[8];
    const float* O1w = (const float*)d_in[9];  const float* O1b = (const float*)d_in[10];
    const float* O2w = (const float*)d_in[11]; const float* O2b = (const float*)d_in[12];
    const float* O3w = (const float*)d_in[13]; const float* O3b = (const float*)d_in[14];
    float* out = (float*)d_out;

    char* ws = (char*)d_ws;
    const size_t HBYTES = (size_t)TOK_ * 256 * sizeof(float);   // 16 MB
    const size_t SBYTES = (size_t)TOK_ * 64  * sizeof(float);   // 4 MB
    float* Ha  = (float*)(ws);
    float* Hb  = (float*)(ws + HBYTES);
    float* Ks  = (float*)(ws + 2 * HBYTES);
    float* Qs  = (float*)(ws + 2 * HBYTES + SBYTES);
    float* WoB = (float*)(ws + 2 * HBYTES + 2 * SBYTES);

    const dim3 blk(256);
    const dim3 gH(TOK_ / 128, 256 / 64);   // hidden layers (out=256)
    const dim3 gO(TOK_ / 128, 1);          // output layers (out=64)

    // Wk path -> Ks = KEY * MLP(KEY)
    mlp_gemm<<<gH, blk, 0, stream>>>(KEY, W1w, W1b, nullptr, Ha, 64, 256, 0);
    mlp_gemm<<<gH, blk, 0, stream>>>(Ha,  W2w, W2b, nullptr, Hb, 256, 256, 0);
    mlp_gemm<<<gO, blk, 0, stream>>>(Hb,  W3w, W3b, KEY,     Ks, 256, 64,  2);
    // Wq path -> Qs = QUERY * MLP(QUERY)
    mlp_gemm<<<gH, blk, 0, stream>>>(QUERY, W1w, W1b, nullptr, Ha, 64, 256, 0);
    mlp_gemm<<<gH, blk, 0, stream>>>(Ha,    W2w, W2b, nullptr, Hb, 256, 256, 0);
    mlp_gemm<<<gO, blk, 0, stream>>>(Hb,    W3w, W3b, QUERY,   Qs, 256, 64,  2);
    // Wo path -> WoB = MLPo(QUERY)
    mlp_gemm<<<gH, blk, 0, stream>>>(QUERY, O1w, O1b, nullptr, Ha, 64, 256, 0);
    mlp_gemm<<<gH, blk, 0, stream>>>(Ha,    O2w, O2b, nullptr, Hb, 256, 256, 0);
    mlp_gemm<<<gO, blk, 0, stream>>>(Hb,    O3w, O3b, nullptr, WoB, 256, 64, 1);

    // fused L1-distance attention
    attn_kernel<<<dim3(NQ_ / 64, B_), blk, 0, stream>>>(Ks, Qs, VALUE, WoB, out);
}
